// MyLSTM_4784593567804
// MI455X (gfx1250) — matrix operand order repaired and checked by reading, not compiled, not run on hardware
//
#include <hip/hip_runtime.h>

// ---------------------------------------------------------------------------
// LSTM  T=8192, I=64, H=2048, O=128  for MI455X (gfx1250, wave32, WMMA)
//
// Phase 1: x_gates[T,4H] = input @ W_ih^T + (b_ih+b_hh)      (bf16 WMMA GEMM)
// Phase 1b: W_hh f32 -> bf16 (packed row-major)               (elementwise)
// Phase 2: persistent-grid recurrence, 256 WGs x 256 threads.
//          Each WG owns 8 h-indices => 32 gate rows of W_hh, held as
//          pre-packed WMMA A-fragments ENTIRELY IN VGPRS (no per-step weight
//          traffic). Per step: h (bf16, 4KB) broadcast via L2, matvec done as
//          16 accumulating v_wmma_f32_16x16x32_bf16 per wave (h in B col 0),
//          gate nonlinearities in fp32, global monotonic-counter barrier.
// Phase 3: out[128] = W_lin @ h_T + b_lin                     (tiny)
// ---------------------------------------------------------------------------

typedef __bf16 v16bf __attribute__((ext_vector_type(16)));
typedef float  v8f   __attribute__((ext_vector_type(8)));

union BFrag {
    v16bf v;
    unsigned short s[16];
    uint4 q[2];
};

__device__ __forceinline__ unsigned short f2bf(float f) {
    unsigned u = __builtin_bit_cast(unsigned, f);
    unsigned r = u + 0x7FFFu + ((u >> 16) & 1u);   // round-to-nearest-even
    return (unsigned short)(r >> 16);
}

#define T_STEPS 8192
#define HID     2048
#define G4      8192      // 4*H
#define NWG     256       // persistent workgroups in phase 2
#define HPB     8         // h-indices per WG (HID / NWG)

// ---------------------------------------------------------------------------
// Phase 1b: convert W_hh (f32, [8192 x 2048]) to bf16, same row-major layout.
// ---------------------------------------------------------------------------
__global__ void lstm_cvt_whh(const float* __restrict__ in,
                             unsigned short* __restrict__ out, int n2) {
    // n2 = number of float PAIRS
    int stride = gridDim.x * blockDim.x;
    for (int i = blockIdx.x * blockDim.x + threadIdx.x; i < n2; i += stride) {
        float2 f = ((const float2*)in)[i];
        unsigned packed = (unsigned)f2bf(f.x) | ((unsigned)f2bf(f.y) << 16);
        ((unsigned*)out)[i] = packed;
    }
}

// ---------------------------------------------------------------------------
// Phase 1: x_gates GEMM. Block = 128 threads (4 waves); wave w computes the
// 16x16 tile (tBase.., gBase + 16w ..). K = 64 => two 16x16x32 bf16 WMMAs.
// C initialized with bias broadcast (C[m][n] = bias[gBase+n]).
// ---------------------------------------------------------------------------
__global__ void lstm_xgates(const float* __restrict__ X,      // [T, 64]
                            const float* __restrict__ Wih,    // [8192, 64]
                            const float* __restrict__ bih,
                            const float* __restrict__ bhh,
                            float* __restrict__ xg) {         // [T, 8192]
    const int lane = threadIdx.x & 31;
    const int wv   = threadIdx.x >> 5;
    const int n    = lane & 15;        // tile row/col index for this lane
    const int hi   = lane >> 4;        // which half of K this lane carries
    const int gBase = blockIdx.x * 64 + wv * 16;
    const int tBase = blockIdx.y * 16;

    const int g = gBase + n;           // gate column this lane owns (B/C/D)
    const float bias = bih[g] + bhh[g];
    v8f acc;
#pragma unroll
    for (int r = 0; r < 8; ++r) acc[r] = bias;

    const int arow = tBase + n;        // input row this lane owns (A)
#pragma unroll
    for (int kb = 0; kb < 64; kb += 32) {
        BFrag a, b;
        // A, this code's contiguous k order: lane(hi=0): K kb+0..15
        //                      lane(hi=1): K kb+16..31
        const float* ap0 = X + (size_t)arow * 64 + kb + hi * 16;
        const float* ap1 = ap0 + 8;
#pragma unroll
        for (int e = 0; e < 8; ++e) {
            a.s[e]     = f2bf(ap0[e]);
            a.s[8 + e] = f2bf(ap1[e]);
        }
        // B, same contiguous k order as A: lane n (hi=0): K kb+0..15 ; (hi=1): K kb+16..31
        const float* bp = Wih + (size_t)g * 64 + kb + hi * 16;
#pragma unroll
        for (int e = 0; e < 16; ++e) b.s[e] = f2bf(bp[e]);

        acc = __builtin_amdgcn_wmma_f32_16x16x32_bf16(
                false, a.v, false, b.v, (short)0, acc, false, false);
    }
    // D: VGPR r, lanes 0-15 -> row tBase+r ; lanes 16-31 -> row tBase+8+r
    float* orow = xg + (size_t)(tBase + hi * 8) * G4 + gBase + n;
#pragma unroll
    for (int r = 0; r < 8; ++r) orow[(size_t)r * G4] = acc[r];
}

// ---------------------------------------------------------------------------
// Phase 2: persistent recurrence.
//   WG b owns h[hBase..hBase+7], gate rows {gt*2048 + hBase + j}.
//   Row-group rg=0 packs [i rows 0..7 | f rows 0..7] as WMMA rows 0..15,
//   rg=1 packs [g | o]. Wave w: rg = w&1, K-quarter kq = w>>1 (512 K each,
//   16 chunks of 32). W fragments live in VGPRs for the whole kernel.
// ---------------------------------------------------------------------------
__global__ void __launch_bounds__(256, 2)
lstm_recurrence(const float* __restrict__ xg,            // [T, 8192]
                const unsigned short* __restrict__ Wbf,  // bf16 [8192, 2048]
                unsigned short* __restrict__ hbuf,       // bf16 [2][2048]
                float* __restrict__ hfinal,              // f32  [2048]
                unsigned* __restrict__ cnt) {
    __shared__ float partial[8][16];
    __shared__ float xgs[32];
    __shared__ float cbuf[HPB];

    const int tid  = threadIdx.x;
    const int lane = tid & 31;
    const int wv   = tid >> 5;
    const int hBase = blockIdx.x * HPB;
    const int rg = wv & 1;
    const int kq = wv >> 1;

    // ---- load this wave's W_hh slice as 16 pre-packed A fragments ----
    const int m  = lane & 15;                // WMMA row this lane carries
    const int hi = lane >> 4;
    const int gt = rg * 2 + (m >> 3);        // gate type 0..3 (i,f,g,o)
    const size_t R = (size_t)(gt * HID + hBase + (m & 7)) * HID;
    v16bf wf[16];
#pragma unroll
    for (int c = 0; c < 16; ++c) {
        const int Kb = kq * 512 + c * 32;
        BFrag f;
        f.q[0] = *(const uint4*)(Wbf + R + Kb + hi * 8);
        f.q[1] = *(const uint4*)(Wbf + R + Kb + 16 + hi * 8);
        wf[c] = f.v;
    }

    // ---- init h0 = 0 (buffer 0), c0 = 0 ----
    if (tid < HPB) {
        cbuf[tid] = 0.0f;
        hbuf[hBase + tid] = 0;               // bf16 zero
    }
    __syncthreads();
    if (tid == 0) {
        __threadfence();                     // release our h0 stores (wave 0)
        __hip_atomic_fetch_add(cnt, 1u, __ATOMIC_RELEASE, __HIP_MEMORY_SCOPE_AGENT);
        while (__hip_atomic_load(cnt, __ATOMIC_ACQUIRE, __HIP_MEMORY_SCOPE_AGENT)
               < (unsigned)NWG)
            __builtin_amdgcn_s_sleep(2);
    }
    __syncthreads();
    __threadfence();                         // acquire: every wave invalidates

    for (int t = 0; t < T_STEPS; ++t) {
        const unsigned short* hprev = hbuf + (t & 1) * HID;

        // prefetch this WG's x_gates slice (32 values: 4 gates x 8 rows)
        if (tid < 32) {
            const int m16 = tid & 15, rgx = tid >> 4;
            const int Rx = (rgx * 2 + (m16 >> 3)) * HID + hBase + (m16 & 7);
            xgs[tid] = xg[(size_t)t * G4 + Rx];
            if (t + 1 < T_STEPS)
                __builtin_prefetch(xg + (size_t)(t + 1) * G4 + Rx, 0, 0);
        }

        // ---- matvec: 16 accumulating WMMAs, h in B column 0 ----
        v8f acc;
#pragma unroll
        for (int r = 0; r < 8; ++r) acc[r] = 0.0f;
#pragma unroll
        for (int c = 0; c < 16; ++c) {
            const int Kb = kq * 512 + c * 32;
            BFrag bb;
            bb.q[0] = make_uint4(0u, 0u, 0u, 0u);
            bb.q[1] = make_uint4(0u, 0u, 0u, 0u);
            if ((lane & 15) == 0) {          // lanes 0 and 16 carry column 0
                const uint4* hp = (const uint4*)(hprev + Kb + hi * 8);
                bb.q[0] = hp[0];
                bb.q[1] = hp[2];
            }
            acc = __builtin_amdgcn_wmma_f32_16x16x32_bf16(
                    false, wf[c], false, bb.v, (short)0, acc, false, false);
        }
        // D column 0: lane 0 -> rows 0..7, lane 16 -> rows 8..15
        if ((lane & 15) == 0) {
#pragma unroll
            for (int r = 0; r < 8; ++r) partial[wv][hi * 8 + r] = acc[r];
        }
        __syncthreads();

        // ---- gate nonlinearities + state update (wave 0, 8 threads) ----
        if (tid < HPB) {
            const int j = tid;
            float iv = xgs[j],      fv = xgs[8 + j];
            float gv = xgs[16 + j], ov = xgs[24 + j];
#pragma unroll
            for (int k = 0; k < 4; ++k) {
                iv += partial[k * 2][j];      fv += partial[k * 2][8 + j];
                gv += partial[k * 2 + 1][j];  ov += partial[k * 2 + 1][8 + j];
            }
            iv = 1.0f / (1.0f + expf(-iv));
            fv = 1.0f / (1.0f + expf(-fv));
            gv = tanhf(gv);
            ov = 1.0f / (1.0f + expf(-ov));
            float cc = fv * cbuf[j] + iv * gv;
            cbuf[j] = cc;
            float hh = ov * tanhf(cc);
            hbuf[((t + 1) & 1) * HID + hBase + j] = f2bf(hh);
            if (t == T_STEPS - 1) hfinal[hBase + j] = hh;
        }
        __syncthreads();

        // ---- grid-wide barrier (monotonic counter). Release: all h stores
        //      came from wave 0, so tid 0's fence orders them before the
        //      signal. Acquire: EVERY thread fences after the wait so all
        //      waves' subsequent hprev loads see fresh data. ----
        if (tid == 0) {
            __threadfence();
            __hip_atomic_fetch_add(cnt, 1u, __ATOMIC_RELEASE, __HIP_MEMORY_SCOPE_AGENT);
            const unsigned tgt = (unsigned)(t + 2) * NWG;
            while (__hip_atomic_load(cnt, __ATOMIC_ACQUIRE, __HIP_MEMORY_SCOPE_AGENT)
                   < tgt)
                __builtin_amdgcn_s_sleep(2);
        }
        __syncthreads();
        __threadfence();                     // acquire for all waves
    }
}

// ---------------------------------------------------------------------------
// Phase 3: out = W_lin @ h_T + b_lin  (128 outputs, trivial)
// ---------------------------------------------------------------------------
__global__ void lstm_linear(const float* __restrict__ Wlin,
                            const float* __restrict__ blin,
                            const float* __restrict__ hfinal,
                            float* __restrict__ out) {
    const int o = threadIdx.x;
    if (o < 128) {
        float s = blin[o];
        const float* wr = Wlin + (size_t)o * HID;
        for (int k = 0; k < HID; ++k) s += wr[k] * hfinal[k];
        out[o] = s;
    }
}

// ---------------------------------------------------------------------------
extern "C" void kernel_launch(void* const* d_in, const int* in_sizes, int n_in,
                              void* d_out, int out_size, void* d_ws, size_t ws_size,
                              hipStream_t stream) {
    const float* X    = (const float*)d_in[0];  // [8192, 64]
    const float* Wih  = (const float*)d_in[1];  // [8192, 64]
    const float* Whh  = (const float*)d_in[2];  // [8192, 2048]
    const float* bih  = (const float*)d_in[3];  // [8192]
    const float* bhh  = (const float*)d_in[4];  // [8192]
    const float* Wlin = (const float*)d_in[5];  // [128, 2048]
    const float* blin = (const float*)d_in[6];  // [128]
    float* out = (float*)d_out;

    char* ws = (char*)d_ws;
    const size_t XG_BYTES  = (size_t)T_STEPS * G4 * sizeof(float);      // 256 MB
    const size_t WBF_BYTES = (size_t)G4 * HID * sizeof(unsigned short); // 32 MB
    float*          xg     = (float*)ws;
    unsigned short* Wbf    = (unsigned short*)(ws + XG_BYTES);
    unsigned short* hbuf   = (unsigned short*)(ws + XG_BYTES + WBF_BYTES);
    float*          hfinal = (float*)(ws + XG_BYTES + WBF_BYTES + 2 * HID * 2);
    unsigned*       cnt    = (unsigned*)(ws + XG_BYTES + WBF_BYTES
                                            + 2 * HID * 2 + HID * 4);

    // barrier counter must start at 0 every call (graph-capturable memset)
    hipMemsetAsync(cnt, 0, 256, stream);

    // Phase 1b: W_hh -> bf16
    lstm_cvt_whh<<<2048, 256, 0, stream>>>(Whh, Wbf, (G4 * HID) / 2);

    // Phase 1: x_gates GEMM (512 T-tiles x 128 gate-quad-tiles, 4 waves/blk)
    lstm_xgates<<<dim3(G4 / 64, T_STEPS / 16), 128, 0, stream>>>(
        X, Wih, bih, bhh, xg);

    // Phase 2: persistent recurrence (requires all 256 WGs co-resident:
    // ~192 VGPRs/wave, ~1KB LDS -> 2 WGs/WGP on a >=128-WGP MI455X)
    lstm_recurrence<<<NWG, 256, 0, stream>>>(xg, Wbf, hbuf, hfinal, cnt);

    // Phase 3: final linear
    lstm_linear<<<1, 128, 0, stream>>>(Wlin, blin, hfinal, out);
}
